// Convolution_506806141527
// MI455X (gfx1250) — matrix operand order repaired and checked by reading, not compiled, not run on hardware
//
#include <hip/hip_runtime.h>
#include <hip/hip_bf16.h>
#include <hip/hip_fp16.h>

// ---------------------------------------------------------------------------
// Fused radial-MLP + tensor-product for MI455X (gfx1250, wave32).
//
// Reformulation: for each weight region the per-edge contraction
//   out[e,w] = sum_u mult[e,u] * ( sum_k h[e,k] * w2[k, col(u,w)] + b2[col] )
// is a single GEMM over fused K=(u,k):
//   out[e,w] = sum_{(u,k)} (h[e,k]*mult[e,u]) * w2[k,col(u,w)]
//            + sum_u mult[e,u] * b2[col(u,w)]
// Both terms run on V_WMMA_F32_16X16X32_F16 with the accumulator chained
// through C->D, so the VALU only builds the A operand (8 packed f16 muls per
// 32-wide K chunk). B tiles stream from LDS with full K=32 occupancy and no
// lane divergence. w2 (f16, 288KB) + b2 transposed (f16, 18KB) + h staging
// (4KB) = 310KB of the 320KB WGP LDS.
//   mult: Wa: x0*sh0 | Wb: x0 | Wc(m): x1[:,m] | Wd: INV_S3*(x1 . sh1)
// ---------------------------------------------------------------------------

typedef __attribute__((ext_vector_type(16))) _Float16 v16h;
typedef __attribute__((ext_vector_type(8)))  _Float16 v8h;
typedef __attribute__((ext_vector_type(8)))  float    v8f;

#define E_EDGES   30000
#define NUM_BASIS 10
#define HIDDEN    16
#define WNUM      9216
#define CUTOFF    4.0f
#define INV_S3    0.57735026918962576f   /* 1/sqrt(3) */
#define PW0       0.10206207261596575f   /* sqrt(1/96) */
#define PW1       0.17677669529663689f   /* sqrt(3/96) */

#define EDGES_PER_WAVE 16
#define WAVES_PER_WG   8
#define EDGES_PER_WG   (EDGES_PER_WAVE * WAVES_PER_WG)   // 128

// LDS layout (dynamic shared), all f16:
//   w2h [col*16 + k]          col<9216, k<16          : 294912 B
//   b2t  region-transposed b2 (see preload)           :  18432 B
//   hst [wave*256 + m*16 + k]                         :   4096 B
#define LDS_W2_HALF  (WNUM * HIDDEN)                 // 147456
#define LDS_W2_BYTES (LDS_W2_HALF * 2)               // 294912
#define LDS_B2_BYTES (WNUM * 2)                      // 18432
#define LDS_H_BYTES  (WAVES_PER_WG * 16 * 16 * 2)    // 4096
#define LDS_BYTES    (LDS_W2_BYTES + LDS_B2_BYTES + LDS_H_BYTES)  // 317440

__device__ __forceinline__ int imin(int a, int b) { return a < b ? a : b; }

__device__ __forceinline__ v8f wmma32(const v16h& A, const v16h& B, const v8f& C) {
  return __builtin_amdgcn_wmma_f32_16x16x32_f16(
      /*neg_a=*/false, A, /*neg_b=*/false, B,
      /*c_mod=*/(short)0, C, /*reuse_a=*/false, /*reuse_b=*/false);
}

// 16 contiguous halves from LDS -> B operand (lane's K slice).
__device__ __forceinline__ v16h lds_b16(const _Float16* __restrict__ p) {
  const v8h* bp = (const v8h*)p;
  v8h lo = bp[0];
  v8h hi = bp[1];
  v16h B;
#pragma unroll
  for (int i = 0; i < 8; ++i) { B[i] = lo[i]; B[i + 8] = hi[i]; }
  return B;
}

// A operand for one fused-K chunk (u0, u0+1):
//  halves 0..7  : h[e][kbase+i] * mult(u0)
//  halves 8..15 : h[e][kbase+i] * mult(u0+1)
__device__ __forceinline__ v16h buildA(const v8h& h8, float f0, float f1) {
  _Float16 m0 = (_Float16)f0;
  _Float16 m1 = (_Float16)f1;
  v16h A;
#pragma unroll
  for (int i = 0; i < 8; ++i) { A[i] = h8[i] * m0; A[i + 8] = h8[i] * m1; }
  return A;
}

__global__ __launch_bounds__(256, 1)
void Convolution_506806141527_kernel(const float* __restrict__ x,
                                     const float* __restrict__ rel,
                                     const float* __restrict__ dist,
                                     const float* __restrict__ freq,
                                     const float* __restrict__ w1,
                                     const float* __restrict__ b1,
                                     const float* __restrict__ w2,
                                     const float* __restrict__ b2,
                                     float* __restrict__ out) {
  extern __shared__ char smem[];
  _Float16* __restrict__ w2h = (_Float16*)smem;
  _Float16* __restrict__ b2t = (_Float16*)(smem + LDS_W2_BYTES);
  _Float16* __restrict__ hst = (_Float16*)(smem + LDS_W2_BYTES + LDS_B2_BYTES);

  const int tid  = threadIdx.x;
  const int wave = tid >> 5;
  const int lane = tid & 31;
  const int half = lane >> 4;   // lane's K half (k 0..7 vs 8..15 in both A and B of the w2 regions)
  const int n    = lane & 15;   // N column within tile; also edge row M for A

  // ---- Stage w2 (16 x 9216 f32) -> LDS f16 [col][k] ----------------------
  for (int idx = tid; idx < LDS_W2_HALF; idx += 256) {
    int k   = idx / WNUM;
    int col = idx - k * WNUM;
    w2h[col * 16 + k] = (_Float16)w2[idx];
  }
  // ---- Stage b2 transposed per region: B-ready layout [w][u] -------------
  for (int idx = tid; idx < WNUM; idx += 256) {
    float val = b2[idx];
    int t;
    if (idx < 4096)      { int u = idx >> 6,          w = idx & 63;          t = w * 64 + u; }
    else if (idx < 6144) { int r = idx - 4096; int u = r >> 5, w = r & 31;   t = 4096 + w * 64 + u; }
    else if (idx < 7168) { int r = idx - 6144; int u = r >> 5, w = r & 31;   t = 6144 + w * 32 + u; }
    else                 { int r = idx - 7168; int u = r >> 6, w = r & 63;   t = 7168 + w * 32 + u; }
    b2t[t] = (_Float16)val;
  }

  // ---- Phase 1: h = silu(basis @ w1 + b1) -> LDS f16 ---------------------
  const int e0 = blockIdx.x * EDGES_PER_WG + wave * EDGES_PER_WAVE;
  if (lane < 16) {
    int e = imin(e0 + lane, E_EDGES - 1);
    float d  = dist[e] * (1.0f / CUTOFF);
    float d5 = d * d * d * d * d;
    float env = 1.0f / d - 28.0f * d5 + 48.0f * d5 * d - 21.0f * d5 * d * d;
    env = (d < 1.0f) ? env : 0.0f;
    float basis[NUM_BASIS];
#pragma unroll
    for (int k = 0; k < NUM_BASIS; ++k) basis[k] = env * __sinf(freq[k] * d);
    _Float16* hrow = hst + wave * 256 + lane * 16;
#pragma unroll
    for (int j = 0; j < HIDDEN; ++j) {
      float h = b1[j];
#pragma unroll
      for (int k = 0; k < NUM_BASIS; ++k) h = fmaf(basis[k], w1[k * HIDDEN + j], h);
      h = h / (1.0f + __expf(-h));
      hrow[j] = (_Float16)h;
    }
  }
  __syncthreads();

  // Per-lane A-row edge (M = lane&15) and its 8 h values (k = half*8 + i).
  v8h h8 = *(const v8h*)(hst + wave * 256 + n * 16 + half * 8);
  const int   ea  = imin(e0 + n, E_EDGES - 1);
  const float* xe = x + (size_t)ea * 160;       // xe[0:64]=x0, xe[64:160]=x1 (32x3)
  __builtin_prefetch(xe, 0, 1);
  const float* re = rel + (size_t)ea * 4;
  const float sh0a  = re[0];
  const float sh1a0 = re[1], sh1a1 = re[2], sh1a2 = re[3];

  v8f acc0[4];    // out0 64 cols (Wa + Wd + their b2 terms)
  v8f accB[2];    // Wb . x0 (32 cols, incl b2b)
  v8f accC[3][2]; // Wc . x1[:,m] (32 cols x 3, incl b2c)
#pragma unroll
  for (int j = 0; j < 4; ++j) acc0[j] = (v8f)(0.0f);
#pragma unroll
  for (int j = 0; j < 2; ++j) {
    accB[j] = (v8f)(0.0f);
#pragma unroll
    for (int m = 0; m < 3; ++m) accC[m][j] = (v8f)(0.0f);
  }

  // ======== Region Wa: cols u*64+w, u<64; mult = x0*sh0 ===================
  for (int c = 0; c < 32; ++c) {
    int u0 = 2 * c;
    v16h Am = buildA(h8, xe[u0] * sh0a, xe[u0 + 1] * sh0a);
    int ub = u0 + half;                       // unused; B elements 0..7 are u0, 8..15 are u0+1
#pragma unroll
    for (int j = 0; j < 4; ++j) {
      const _Float16* bq = w2h + (unsigned)(u0 * 64 + j * 16 + n) * 16 + half * 8;
      v8h lo = *(const v8h*)bq, hi = *(const v8h*)(bq + 64 * 16);
      v16h B;
      for (int i = 0; i < 8; ++i) { B[i] = lo[i]; B[i + 8] = hi[i]; }
      acc0[j] = wmma32(Am, B, acc0[j]);
    }
  }

  // ======== Region Wb: cols 4096 + u*32+w, u<64; mult = x0 ================
  for (int c = 0; c < 32; ++c) {
    int u0 = 2 * c;
    v16h Am = buildA(h8, xe[u0], xe[u0 + 1]);
    int ub = u0 + half;
#pragma unroll
    for (int j = 0; j < 2; ++j) {
      const _Float16* bq = w2h + (unsigned)(4096 + u0 * 32 + j * 16 + n) * 16 + half * 8;
      v8h lo = *(const v8h*)bq, hi = *(const v8h*)(bq + 32 * 16);
      v16h B;
      for (int i = 0; i < 8; ++i) { B[i] = lo[i]; B[i + 8] = hi[i]; }
      accB[j] = wmma32(Am, B, accB[j]);
    }
  }

  // ======== Region Wc: cols 6144 + u*32+w, u<32; mult_m = x1[u][m] ========
  for (int c = 0; c < 16; ++c) {
    int u0 = 2 * c;
    const float* p = xe + 64 + u0 * 3;        // x1[u0][0..2], x1[u0+1][0..2]
    v16h Am0 = buildA(h8, p[0], p[3]);
    v16h Am1 = buildA(h8, p[1], p[4]);
    v16h Am2 = buildA(h8, p[2], p[5]);
    int ub = u0 + half;
#pragma unroll
    for (int j = 0; j < 2; ++j) {
      const _Float16* bq = w2h + (unsigned)(6144 + u0 * 32 + j * 16 + n) * 16 + half * 8;
      v8h lo = *(const v8h*)bq, hi = *(const v8h*)(bq + 32 * 16);
      v16h B;
      for (int i = 0; i < 8; ++i) { B[i] = lo[i]; B[i + 8] = hi[i]; }
      accC[0][j] = wmma32(Am0, B, accC[0][j]);
      accC[1][j] = wmma32(Am1, B, accC[1][j]);
      accC[2][j] = wmma32(Am2, B, accC[2][j]);
    }
  }

  // ======== Region Wd: cols 7168 + u*64+w, u<32; mult = INV_S3*(x1.sh1) ===
  for (int c = 0; c < 16; ++c) {
    int u0 = 2 * c;
    const float* p = xe + 64 + u0 * 3;
    float t0 = INV_S3 * (p[0] * sh1a0 + p[1] * sh1a1 + p[2] * sh1a2);
    float t1 = INV_S3 * (p[3] * sh1a0 + p[4] * sh1a1 + p[5] * sh1a2);
    v16h Am = buildA(h8, t0, t1);
    int ub = u0 + half;
#pragma unroll
    for (int j = 0; j < 4; ++j) {
      const _Float16* bq = w2h + (unsigned)(7168 + u0 * 64 + j * 16 + n) * 16 + half * 8;
      v8h lo = *(const v8h*)bq, hi = *(const v8h*)(bq + 64 * 16);
      v16h B;
      for (int i = 0; i < 8; ++i) { B[i] = lo[i]; B[i + 8] = hi[i]; }
      acc0[j] = wmma32(Am, B, acc0[j]);
    }
  }

  // ======== b2 terms: GEMM with K = u against transposed b2 in LDS ========
  // b2a: K=64 (2 chunks), mult = x0*sh0, -> acc0
#pragma unroll
  for (int c = 0; c < 2; ++c) {
    v16h Am;
#pragma unroll
    for (int i = 0; i < 8; ++i) {
      int uA = c * 32 + half * 16 + i;
      int uB = c * 32 + half * 16 + 8 + i;
      Am[i]     = (_Float16)(xe[uA] * sh0a);
      Am[i + 8] = (_Float16)(xe[uB] * sh0a);
    }
#pragma unroll
    for (int j = 0; j < 4; ++j) {
      v16h B = lds_b16(b2t + (unsigned)((j * 16 + n) * 64 + c * 32 + half * 16));
      acc0[j] = wmma32(Am, B, acc0[j]);
    }
  }
  // b2b: K=64 (2 chunks), mult = x0, -> accB
#pragma unroll
  for (int c = 0; c < 2; ++c) {
    v16h Am;
#pragma unroll
    for (int i = 0; i < 8; ++i) {
      Am[i]     = (_Float16)xe[c * 32 + half * 16 + i];
      Am[i + 8] = (_Float16)xe[c * 32 + half * 16 + 8 + i];
    }
#pragma unroll
    for (int j = 0; j < 2; ++j) {
      v16h B = lds_b16(b2t + (unsigned)(4096 + (j * 16 + n) * 64 + c * 32 + half * 16));
      accB[j] = wmma32(Am, B, accB[j]);
    }
  }
  // b2c: K=32 (1 chunk), mult_m = x1[u][m], -> accC
  {
    v16h Am[3];
#pragma unroll
    for (int m = 0; m < 3; ++m) {
#pragma unroll
      for (int i = 0; i < 8; ++i) {
        int uA = half * 16 + i;
        int uB = half * 16 + 8 + i;
        Am[m][i]     = (_Float16)xe[64 + uA * 3 + m];
        Am[m][i + 8] = (_Float16)xe[64 + uB * 3 + m];
      }
    }
#pragma unroll
    for (int j = 0; j < 2; ++j) {
      v16h B = lds_b16(b2t + (unsigned)(6144 + (j * 16 + n) * 32 + half * 16));
      accC[0][j] = wmma32(Am[0], B, accC[0][j]);
      accC[1][j] = wmma32(Am[1], B, accC[1][j]);
      accC[2][j] = wmma32(Am[2], B, accC[2][j]);
    }
  }
  // b2d: K=32 (1 chunk), mult = INV_S3*(x1.sh1), -> acc0
  {
    v16h Am;
#pragma unroll
    for (int i = 0; i < 8; ++i) {
      int uA = half * 16 + i;
      int uB = half * 16 + 8 + i;
      const float* pA = xe + 64 + uA * 3;
      const float* pB = xe + 64 + uB * 3;
      Am[i]     = (_Float16)(INV_S3 * (pA[0] * sh1a0 + pA[1] * sh1a1 + pA[2] * sh1a2));
      Am[i + 8] = (_Float16)(INV_S3 * (pB[0] * sh1a0 + pB[1] * sh1a1 + pB[2] * sh1a2));
    }
#pragma unroll
    for (int j = 0; j < 4; ++j) {
      v16h B = lds_b16(b2t + (unsigned)(7168 + (j * 16 + n) * 32 + half * 16));
      acc0[j] = wmma32(Am, B, acc0[j]);
    }
  }

  // ======== Epilogue: lane holds (M = v + 8*half, N = n) ==================
#pragma unroll
  for (int v = 0; v < 8; ++v) {
    int e = e0 + v + 8 * half;
    if (e < E_EDGES) {
      const float* r = rel + (size_t)e * 4;
      float s0 = r[0], s1 = r[1], s2 = r[2], s3 = r[3];
      float* orow = out + (size_t)e * 160;
#pragma unroll
      for (int j = 0; j < 4; ++j) orow[j * 16 + n] = PW0 * acc0[j][v];
#pragma unroll
      for (int j = 0; j < 2; ++j) {
        int w = j * 16 + n;
        float sb = accB[j][v];
        orow[64 + 3 * w + 0] = (PW1 * INV_S3) * (sb * s1 + accC[0][j][v] * s0);
        orow[64 + 3 * w + 1] = (PW1 * INV_S3) * (sb * s2 + accC[1][j][v] * s0);
        orow[64 + 3 * w + 2] = (PW1 * INV_S3) * (sb * s3 + accC[2][j][v] * s0);
      }
    }
  }
}

extern "C" void kernel_launch(void* const* d_in, const int* in_sizes, int n_in,
                              void* d_out, int out_size, void* d_ws, size_t ws_size,
                              hipStream_t stream) {
  const float* x    = (const float*)d_in[0];
  const float* rel  = (const float*)d_in[1];
  const float* dist = (const float*)d_in[2];
  const float* freq = (const float*)d_in[3];
  const float* w1   = (const float*)d_in[4];
  const float* b1   = (const float*)d_in[5];
  const float* w2   = (const float*)d_in[6];
  const float* b2   = (const float*)d_in[7];
  float* out = (float*)d_out;

  (void)hipFuncSetAttribute((const void*)Convolution_506806141527_kernel,
                            hipFuncAttributeMaxDynamicSharedMemorySize, LDS_BYTES);

  int grid = (E_EDGES + EDGES_PER_WG - 1) / EDGES_PER_WG;  // 235
  Convolution_506806141527_kernel<<<grid, 256, LDS_BYTES, stream>>>(
      x, rel, dist, freq, w1, b1, w2, b2, out);
}
